// GPT_35579509080716
// MI455X (gfx1250) — compile-verified, matrix operand order checked by reading, not run on hardware
//
#include <hip/hip_runtime.h>

typedef __bf16 bf16;
typedef __bf16 v16bf __attribute__((ext_vector_type(16)));
typedef __bf16 v8bf  __attribute__((ext_vector_type(8)));
typedef float  v8f   __attribute__((ext_vector_type(8)));

#define B_  2
#define S_  1024
#define D_  768
#define H_  12
#define HD_ 64
#define L_  6
#define V_  32000
#define M_  (B_ * S_)

// LDS tile row stride in elements (32 data + 8 pad -> 80B row stride,
// bank step 20 mod 64, conflict-free b128 reads across 16 rows)
#define LLD 40

#if defined(__has_builtin)
#if __has_builtin(__builtin_amdgcn_global_load_async_to_lds_b128)
#define HAVE_ASYNC_LDS 1
#endif
#endif

// ---------------------------------------------------------------------------
// 16B global -> LDS copy. Async (ASYNCcnt-tracked) when available.
// Builtin signature (from hipcc diagnostic): param0 is AS1 pointer to int
// __vector(4); cast to exact types (C-style cast = addrspacecast + bitcast).
// ---------------------------------------------------------------------------
#if defined(HAVE_ASYNC_LDS)
typedef int i32x4_ __attribute__((ext_vector_type(4)));
typedef __attribute__((address_space(1))) i32x4_* g4p_t;
typedef __attribute__((address_space(3))) i32x4_* l4p_t;
#endif

__device__ __forceinline__ void copy16_g2l(const bf16* g, bf16* l) {
#if defined(HAVE_ASYNC_LDS)
  __builtin_amdgcn_global_load_async_to_lds_b128((g4p_t)g, (l4p_t)l, 0, 0);
#else
  *(v8bf*)l = *(const v8bf*)g;
#endif
}

__device__ __forceinline__ void wait_async_fills() {
#if defined(HAVE_ASYNC_LDS)
#if __has_builtin(__builtin_amdgcn_s_wait_asynccnt)
  __builtin_amdgcn_s_wait_asynccnt(0);
#else
  asm volatile("s_wait_asynccnt 0x0" ::: "memory");
#endif
#endif
}

// ---------------------------------------------------------------------------
// WMMA fragment load from an LDS tile (row-major, leading dim LLD).
// CDNA5 16-bit A/B layout: lanes 0-15 m=lane, elems {0..7,8..15} = K {0..7,16..23};
// lanes 16-31: K {8..15,24..31}.  => two 16B chunks at +hi*16 and +hi*16+32 bytes.
// ---------------------------------------------------------------------------
__device__ __forceinline__ v16bf frag_lds(const bf16* tile, int row, int lane) {
  int r  = lane & 15;
  int hi = lane >> 4;
  const bf16* p = tile + (row + r) * LLD + hi * 8;
  v8bf lo  = *(const v8bf*)(p);
  v8bf hi8 = *(const v8bf*)(p + 16);
  return __builtin_shufflevector(lo, hi8, 0, 1, 2, 3, 4, 5, 6, 7,
                                 8, 9, 10, 11, 12, 13, 14, 15);
}

// ---------------------------------------------------------------------------
// Generic batched GEMM:  C[z] = epilogue( alpha * A[z] @ W[z]^T + bias )
//   A : bf16 [M x K] row-major, leading dim lda, batch stride sA
//   W : bf16 [N x K] row-major (weight already transposed), ldw, stride sW
//   C batch-z base: (z/batchH)*cOuter + (z%batchH)*cInner
// flags: bit0 = causal mask (col>row -> -1e30), bit1 = exact GELU
// Block: 256 threads (8 waves), 128x128 C tile, K-step 32, LDS double-buffered
// async staging.
// ---------------------------------------------------------------------------
__global__ __launch_bounds__(256) void k_gemm(
    const bf16* __restrict__ A, const bf16* __restrict__ W,
    const float* __restrict__ bias, const float* __restrict__ res,
    float* __restrict__ outF, bf16* __restrict__ out16,
    int Mz, int N, int K, int lda, int ldw, int ldc,
    long long sA, long long sW, long long cOuter, long long cInner, int batchH,
    float alpha, int flags)
{
  __shared__ __align__(16) bf16 sAt[2][128 * LLD];
  __shared__ __align__(16) bf16 sBt[2][128 * LLD];

  int z = blockIdx.z;
  A += (size_t)z * sA;
  W += (size_t)z * sW;
  size_t cbase = (size_t)(z / batchH) * (size_t)cOuter +
                 (size_t)(z % batchH) * (size_t)cInner;

  int lane = threadIdx.x & 31;
  int wave = threadIdx.x >> 5;
  int wm = wave & 3;   // 4 waves along M -> 32 rows each
  int wn = wave >> 2;  // 2 waves along N -> 64 cols each
  int rowBlk = blockIdx.y * 128;
  int colBlk = blockIdx.x * 128;

  // Fill one K-step (32 cols) of both tiles: 128 rows x 4 chunks of 16B each,
  // 512 chunks per tile, 2 per thread. Uniform control flow (EXEC stays ~0xffffffff).
  auto fill = [&](int buf, int k0) {
#pragma unroll
    for (int c = (int)threadIdx.x; c < 512; c += 256) {
      int r = c >> 2, seg = c & 3;
      copy16_g2l(A + (size_t)(rowBlk + r) * lda + k0 + seg * 8,
                 &sAt[buf][r * LLD + seg * 8]);
      int rb = colBlk + r; rb = rb < N ? rb : (N - 1);
      copy16_g2l(W + (size_t)rb * ldw + k0 + seg * 8,
                 &sBt[buf][r * LLD + seg * 8]);
    }
  };

  v8f acc[2][4] = {};

  int nk = K >> 5;  // K is always a multiple of 32 here
  fill(0, 0);
  for (int it = 0; it < nk; it++) {
    int cur = it & 1;
    wait_async_fills();   // my fills for buf[cur] are done
    __syncthreads();      // everyone's fills done; prev compute done everywhere
    if (it + 1 < nk) fill(cur ^ 1, (it + 1) << 5);

    v16bf af[2];
    v16bf bf_[4];
#pragma unroll
    for (int i = 0; i < 2; i++)
      af[i] = frag_lds(sAt[cur], wm * 32 + i * 16, lane);
#pragma unroll
    for (int j = 0; j < 4; j++)
      bf_[j] = frag_lds(sBt[cur], wn * 64 + j * 16, lane);
#pragma unroll
    for (int i = 0; i < 2; i++)
#pragma unroll
      for (int j = 0; j < 4; j++)
        acc[i][j] = __builtin_amdgcn_wmma_f32_16x16x32_bf16(
            false, af[i], false, bf_[j], (short)0, acc[i][j], false, false);
  }

  // C/D layout: lane n = lane&15; VGPR r -> row = (lane>=16 ? 8 : 0) + r
  int n   = lane & 15;
  int mro = (lane >> 4) * 8;
#pragma unroll
  for (int i = 0; i < 2; i++) {
#pragma unroll
    for (int j = 0; j < 4; j++) {
      int col = colBlk + wn * 64 + j * 16 + n;
      if (col >= N) continue;
      float bv = bias ? bias[col] : 0.0f;
#pragma unroll
      for (int r = 0; r < 8; r++) {
        int row = rowBlk + wm * 32 + i * 16 + mro + r;
        if (row >= Mz) continue;
        float v = acc[i][j][r] * alpha + bv;
        if (flags & 1) { if (col > row) v = -1.0e30f; }
        if (flags & 2) { v = 0.5f * v * (1.0f + erff(v * 0.70710678118654752f)); }
        size_t cidx = cbase + (size_t)row * ldc + col;
        if (res)   v += res[cidx];
        if (outF)  outF[cidx]  = v;
        if (out16) out16[cidx] = (bf16)v;
      }
    }
  }
}

// ---------------------------------------------------------------------------
// LayerNorm over D=768, one block per row, bf16 output.
// ---------------------------------------------------------------------------
__global__ __launch_bounds__(256) void k_layernorm(
    const float* __restrict__ x, const float* __restrict__ g,
    const float* __restrict__ b, bf16* __restrict__ out)
{
  __shared__ float red[256];
  int row = blockIdx.x;
  int t = threadIdx.x;
  const float* xr = x + (size_t)row * D_;
  float v0 = xr[t], v1 = xr[t + 256], v2 = xr[t + 512];
  red[t] = v0 + v1 + v2;
  __syncthreads();
  for (int off = 128; off > 0; off >>= 1) {
    if (t < off) red[t] += red[t + off];
    __syncthreads();
  }
  float mu = red[0] * (1.0f / D_);
  __syncthreads();
  float d0 = v0 - mu, d1 = v1 - mu, d2 = v2 - mu;
  red[t] = d0 * d0 + d1 * d1 + d2 * d2;
  __syncthreads();
  for (int off = 128; off > 0; off >>= 1) {
    if (t < off) red[t] += red[t + off];
    __syncthreads();
  }
  float rs = rsqrtf(red[0] * (1.0f / D_) + 1e-5f);
  bf16* o = out + (size_t)row * D_;
  o[t]       = (bf16)(d0 * rs * g[t]       + b[t]);
  o[t + 256] = (bf16)(d1 * rs * g[t + 256] + b[t + 256]);
  o[t + 512] = (bf16)(d2 * rs * g[t + 512] + b[t + 512]);
}

// ---------------------------------------------------------------------------
// Row softmax over S=1024 (att fp32 -> probs bf16).
// ---------------------------------------------------------------------------
__global__ __launch_bounds__(256) void k_softmax(const float* __restrict__ att,
                                                 bf16* __restrict__ p)
{
  __shared__ float red[256];
  size_t row = blockIdx.x;
  int t = threadIdx.x;
  const float* a = att + row * S_;
  float v[4];
#pragma unroll
  for (int i = 0; i < 4; i++) v[i] = a[t + i * 256];
  float mx = fmaxf(fmaxf(v[0], v[1]), fmaxf(v[2], v[3]));
  red[t] = mx;
  __syncthreads();
  for (int off = 128; off > 0; off >>= 1) {
    if (t < off) red[t] = fmaxf(red[t], red[t + off]);
    __syncthreads();
  }
  mx = red[0];
  __syncthreads();
  float s = 0.0f;
#pragma unroll
  for (int i = 0; i < 4; i++) { v[i] = expf(v[i] - mx); s += v[i]; }
  red[t] = s;
  __syncthreads();
  for (int off = 128; off > 0; off >>= 1) {
    if (t < off) red[t] += red[t + off];
    __syncthreads();
  }
  float inv = 1.0f / red[0];
  bf16* o = p + row * S_;
#pragma unroll
  for (int i = 0; i < 4; i++) o[t + i * 256] = (bf16)(v[i] * inv);
}

// ---------------------------------------------------------------------------
// Weight fp32 [K,N] -> bf16 transposed [N,K]
// ---------------------------------------------------------------------------
__global__ void k_w_transpose_bf16(const float* __restrict__ W,
                                   bf16* __restrict__ Wt, int K, int N)
{
  size_t t = (size_t)blockIdx.x * blockDim.x + threadIdx.x;
  if (t >= (size_t)K * N) return;
  int k = (int)(t % K);
  int n = (int)(t / K);
  Wt[(size_t)n * K + k] = (bf16)W[(size_t)k * N + n];
}

__global__ void k_f2b(const float* __restrict__ src, bf16* __restrict__ dst, size_t n)
{
  size_t t = (size_t)blockIdx.x * blockDim.x + threadIdx.x;
  if (t < n) dst[t] = (bf16)src[t];
}

// ---------------------------------------------------------------------------
// x = tok_emb[idx] + pos_emb
// ---------------------------------------------------------------------------
__global__ void k_embed(const int* __restrict__ idx, const float* __restrict__ tok,
                        const float* __restrict__ pos, float* __restrict__ x)
{
  size_t t = (size_t)blockIdx.x * blockDim.x + threadIdx.x;
  if (t >= (size_t)M_ * D_) return;
  int d = (int)(t % D_);
  int row = (int)(t / D_);
  int s = row & (S_ - 1);
  x[t] = tok[(size_t)idx[row] * D_ + d] + pos[(size_t)s * D_ + d];
}

// ---------------------------------------------------------------------------
// qkv [M,3D] fp32 -> q,k bf16 [bh][S][HD] and v^T bf16 [bh][HD][S]
// ---------------------------------------------------------------------------
__global__ void k_split_heads(const float* __restrict__ qkv, bf16* __restrict__ q,
                              bf16* __restrict__ k, bf16* __restrict__ vt)
{
  size_t t = (size_t)blockIdx.x * blockDim.x + threadIdx.x;
  size_t total = (size_t)B_ * H_ * S_ * HD_;
  if (t >= total) return;
  int d = (int)(t % HD_);
  int s = (int)((t / HD_) % S_);
  int z = (int)(t / ((size_t)HD_ * S_));
  int b = z / H_, h = z % H_;
  size_t base = ((size_t)(b * S_ + s)) * (3 * D_) + h * HD_ + d;
  q[t] = (bf16)qkv[base];
  k[t] = (bf16)qkv[base + D_];
  vt[(size_t)z * HD_ * S_ + (size_t)d * S_ + s] = (bf16)qkv[base + 2 * D_];
}

// ---------------------------------------------------------------------------
static inline void gemm(hipStream_t st, const bf16* A, const bf16* W,
                        const float* bias, const float* res, float* outF, bf16* o16,
                        int M, int N, int K, int lda, int ldw, int ldc,
                        long long sA, long long sW, long long cO, long long cI,
                        int batchH, int batch, float alpha, int flags)
{
  dim3 g((N + 127) / 128, (M + 127) / 128, batch);
  k_gemm<<<g, 256, 0, st>>>(A, W, bias, res, outF, o16, M, N, K, lda, ldw, ldc,
                            sA, sW, cO, cI, batchH, alpha, flags);
}

extern "C" void kernel_launch(void* const* d_in, const int* in_sizes, int n_in,
                              void* d_out, int out_size, void* d_ws, size_t ws_size,
                              hipStream_t stream)
{
  const int*   idx  = (const int*)d_in[0];
  const float* tok  = (const float*)d_in[1];
  const float* pos  = (const float*)d_in[2];
  const float* ln1g = (const float*)d_in[3];
  const float* ln1b = (const float*)d_in[4];
  const float* qkvw = (const float*)d_in[5];
  const float* qkvb = (const float*)d_in[6];
  const float* attw = (const float*)d_in[7];
  const float* attb = (const float*)d_in[8];
  const float* ln2g = (const float*)d_in[9];
  const float* ln2b = (const float*)d_in[10];
  const float* fcw  = (const float*)d_in[11];
  const float* fcb  = (const float*)d_in[12];
  const float* mlpw = (const float*)d_in[13];
  const float* mlpb = (const float*)d_in[14];
  const float* lnfg = (const float*)d_in[15];
  const float* lnfb = (const float*)d_in[16];
  float* out = (float*)d_out;

  char* wsb = (char*)d_ws;
  size_t off = 0;
  auto carve = [&](size_t bytes) -> char* {
    char* p = wsb + off;
    off += (bytes + 255) & ~(size_t)255;
    return p;
  };

  float* x     = (float*)carve((size_t)M_ * D_ * 4);
  bf16*  ln16  = (bf16*) carve((size_t)M_ * D_ * 2);
  float* qkvf  = (float*)carve((size_t)M_ * 3 * D_ * 4);
  bf16*  qh    = (bf16*) carve((size_t)B_ * H_ * S_ * HD_ * 2);
  bf16*  kh    = (bf16*) carve((size_t)B_ * H_ * S_ * HD_ * 2);
  bf16*  vt    = (bf16*) carve((size_t)B_ * H_ * S_ * HD_ * 2);
  float* att   = (float*)carve((size_t)B_ * H_ * S_ * S_ * 4);
  bf16*  att16 = (bf16*) carve((size_t)B_ * H_ * S_ * S_ * 2);
  bf16*  y16   = (bf16*) carve((size_t)M_ * D_ * 2);
  bf16*  fc16  = (bf16*) carve((size_t)M_ * 4 * D_ * 2);
  bf16*  wqT   = (bf16*) carve((size_t)L_ * 3 * D_ * D_ * 2);
  bf16*  waT   = (bf16*) carve((size_t)L_ * D_ * D_ * 2);
  bf16*  wfT   = (bf16*) carve((size_t)L_ * 4 * D_ * D_ * 2);
  bf16*  wmT   = (bf16*) carve((size_t)L_ * 4 * D_ * D_ * 2);
  bf16*  wemb  = (bf16*) carve((size_t)V_ * D_ * 2);

  // ---- weight conversion (fp32 -> bf16 N x K) --------------------------------
  for (int l = 0; l < L_; l++) {
    {
      size_t n = (size_t)D_ * 3 * D_;
      k_w_transpose_bf16<<<(unsigned)((n + 255) / 256), 256, 0, stream>>>(
          qkvw + (size_t)l * n, wqT + (size_t)l * n, D_, 3 * D_);
    }
    {
      size_t n = (size_t)D_ * D_;
      k_w_transpose_bf16<<<(unsigned)((n + 255) / 256), 256, 0, stream>>>(
          attw + (size_t)l * n, waT + (size_t)l * n, D_, D_);
    }
    {
      size_t n = (size_t)D_ * 4 * D_;
      k_w_transpose_bf16<<<(unsigned)((n + 255) / 256), 256, 0, stream>>>(
          fcw + (size_t)l * n, wfT + (size_t)l * n, D_, 4 * D_);
    }
    {
      size_t n = (size_t)4 * D_ * D_;
      k_w_transpose_bf16<<<(unsigned)((n + 255) / 256), 256, 0, stream>>>(
          mlpw + (size_t)l * n, wmT + (size_t)l * n, 4 * D_, D_);
    }
  }
  {
    size_t n = (size_t)V_ * D_;  // tok_emb already [V, D] == N x K for lm_head
    k_f2b<<<(unsigned)((n + 255) / 256), 256, 0, stream>>>(tok, wemb, n);
  }

  // ---- embedding -------------------------------------------------------------
  {
    size_t n = (size_t)M_ * D_;
    k_embed<<<(unsigned)((n + 255) / 256), 256, 0, stream>>>(idx, tok, pos, x);
  }

  const int BH = B_ * H_;
  // ---- transformer layers ----------------------------------------------------
  for (int l = 0; l < L_; l++) {
    k_layernorm<<<M_, 256, 0, stream>>>(x, ln1g + l * D_, ln1b + l * D_, ln16);

    // qkv = ln16 @ qkv_w + qkv_b
    gemm(stream, ln16, wqT + (size_t)l * 3 * D_ * D_, qkvb + l * 3 * D_,
         nullptr, qkvf, nullptr,
         M_, 3 * D_, D_, D_, D_, 3 * D_, 0, 0, 0, 0, 1, 1, 1.0f, 0);

    {
      size_t n = (size_t)BH * S_ * HD_;
      k_split_heads<<<(unsigned)((n + 255) / 256), 256, 0, stream>>>(qkvf, qh, kh, vt);
    }

    // att = scale * q @ k^T  (causal mask in epilogue), batched over b*h
    gemm(stream, qh, kh, nullptr, nullptr, att, nullptr,
         S_, S_, HD_, HD_, HD_, S_,
         (long long)S_ * HD_, (long long)S_ * HD_, (long long)S_ * S_, 0, 1,
         BH, 0.125f, 1);

    k_softmax<<<BH * S_, 256, 0, stream>>>(att, att16);

    // y[b,s,h*64+d] = att @ v   (write directly interleaved, bf16)
    gemm(stream, att16, vt, nullptr, nullptr, nullptr, y16,
         S_, HD_, S_, S_, S_, D_,
         (long long)S_ * S_, (long long)HD_ * S_,
         (long long)S_ * D_, (long long)HD_, H_, BH, 1.0f, 0);

    // x = x + y @ attn_w + attn_b   (in-place residual)
    gemm(stream, y16, waT + (size_t)l * D_ * D_, attb + l * D_,
         x, x, nullptr,
         M_, D_, D_, D_, D_, D_, 0, 0, 0, 0, 1, 1, 1.0f, 0);

    k_layernorm<<<M_, 256, 0, stream>>>(x, ln2g + l * D_, ln2b + l * D_, ln16);

    // h = gelu(ln16 @ fc_w + fc_b) -> bf16
    gemm(stream, ln16, wfT + (size_t)l * 4 * D_ * D_, fcb + l * 4 * D_,
         nullptr, nullptr, fc16,
         M_, 4 * D_, D_, D_, D_, 4 * D_, 0, 0, 0, 0, 1, 1, 1.0f, 2);

    // x = x + h @ mlp_w + mlp_b   (in-place residual)
    gemm(stream, fc16, wmT + (size_t)l * 4 * D_ * D_, mlpb + l * D_,
         x, x, nullptr,
         M_, D_, 4 * D_, 4 * D_, 4 * D_, D_, 0, 0, 0, 0, 1, 1, 1.0f, 0);
  }

  // ---- final LN + tied lm_head ----------------------------------------------
  k_layernorm<<<M_, 256, 0, stream>>>(x, lnfg, lnfb, ln16);

  gemm(stream, ln16, wemb, nullptr, nullptr, out, nullptr,
       M_, V_, D_, D_, D_, V_, 0, 0, 0, 0, 1, 1, 1.0f, 0);
}
